// TinyVitAttention_66262755442832
// MI455X (gfx1250) — matrix operand order repaired and checked by reading, not compiled, not run on hardware
//
#include <hip/hip_runtime.h>
#include <hip/hip_bf16.h>

typedef __attribute__((ext_vector_type(16))) _Float16 v16h;
typedef __attribute__((ext_vector_type(8)))  _Float16 v8h;
typedef __attribute__((ext_vector_type(8)))  float    v8f;
typedef int b128_t __attribute__((__vector_size__(16)));   // async-LDS builtin's unit

#define TOKENS 25088      // 128 * 196
#define SEQ    196
#define SEQP   224        // SEQ padded to 7*32 for ctx K-loop
#define KTP    208        // SEQ padded to 13*16 for kT buffer
#define NHEADS 18
#define KD     32         // key dim (== WMMA K for f16)
#define VD     128
#define HQKV   3456
#define DHID   2304
#define DIMC   576
#define QTILES 13         // ceil(196/16)

union Frag {
    v16h v;
    v8h  h2[2];
    _Float16 e[16];
};

// ---------------------------------------------------------------------------
// gfx1250 async global->LDS helpers (guarded; fallback = synchronous copy)
// ---------------------------------------------------------------------------
#if defined(__HIP_DEVICE_COMPILE__) && __has_builtin(__builtin_amdgcn_global_load_async_to_lds_b128)
#define HAVE_ASYNC_LDS 1
#else
#define HAVE_ASYNC_LDS 0
#endif

__device__ __forceinline__ void async_copy_b128(const _Float16* g, _Float16* l)
{
#if HAVE_ASYNC_LDS
    void* gv = (void*)g;   // drop const
    void* lv = (void*)l;
    __builtin_amdgcn_global_load_async_to_lds_b128(
        (__attribute__((address_space(1))) b128_t*)gv,
        (__attribute__((address_space(3))) b128_t*)lv,
        0, 0);
#else
    *(v8h*)l = *(const v8h*)g;
#endif
}

__device__ __forceinline__ void async_wait_all()
{
#if HAVE_ASYNC_LDS
#if __has_builtin(__builtin_amdgcn_s_wait_asynccnt)
    __builtin_amdgcn_s_wait_asynccnt(0);
#else
    asm volatile("s_wait_asynccnt 0x0" ::: "memory");
#endif
#endif
}

// wave-local LDS ordering fence (LDS is in-order per wave; this stops the
// compiler from reordering and drains DScnt before cross-lane LDS reads)
__device__ __forceinline__ void wave_lds_fence()
{
    asm volatile("s_wait_dscnt 0x0" ::: "memory");
}

// ---------------------------------------------------------------------------
// weight convert + transpose: src f32 [Nr][K] -> dst f16 [K][Nr]
// ---------------------------------------------------------------------------
__global__ void __launch_bounds__(256)
cvt_transpose_f16(const float* __restrict__ src, _Float16* __restrict__ dst,
                  int Nr, int K)
{
    int i = blockIdx.x * 256 + threadIdx.x;
    if (i < Nr * K) {
        int n = i / K, k = i % K;
        dst[(size_t)k * Nr + n] = (_Float16)src[i];
    }
}

// ---------------------------------------------------------------------------
// LayerNorm over DIM=576 per token, output f16
// ---------------------------------------------------------------------------
__global__ void __launch_bounds__(256)
layernorm_f16(const float* __restrict__ x, const float* __restrict__ w,
              const float* __restrict__ bset, _Float16* __restrict__ xh)
{
    __shared__ float red[256];
    const int t = blockIdx.x;
    const float* row = x + (size_t)t * DIMC;
    float s = 0.f, s2 = 0.f;
    for (int i = threadIdx.x; i < DIMC; i += 256) {
        float v = row[i];
        s += v; s2 += v * v;
    }
    red[threadIdx.x] = s;
    __syncthreads();
    for (int o = 128; o > 0; o >>= 1) {
        if (threadIdx.x < o) red[threadIdx.x] += red[threadIdx.x + o];
        __syncthreads();
    }
    float mean = red[0] * (1.f / DIMC);
    __syncthreads();
    red[threadIdx.x] = s2;
    __syncthreads();
    for (int o = 128; o > 0; o >>= 1) {
        if (threadIdx.x < o) red[threadIdx.x] += red[threadIdx.x + o];
        __syncthreads();
    }
    float var = red[0] * (1.f / DIMC) - mean * mean;
    float rs = rsqrtf(var + 1e-5f);
    for (int i = threadIdx.x; i < DIMC; i += 256)
        xh[(size_t)t * DIMC + i] = (_Float16)((row[i] - mean) * rs * w[i] + bset[i]);
}

// ---------------------------------------------------------------------------
// C[M,N] = A[M,K] @ Wt[K,N] + bias, one wave per 16x16 tile, all-vector frags.
// mode 0: store f32 to outF32[M][N]
// mode 1: QKV scatter: q -> qh[tok][18][32], k -> kT[b][h][32][208] (transposed),
//         v -> vh[tok][18][128], all f16
// ---------------------------------------------------------------------------
__global__ void __launch_bounds__(256)
gemm_f16_wmma(const _Float16* __restrict__ A, const _Float16* __restrict__ Wt,
              const float* __restrict__ bias, float* __restrict__ outF32,
              _Float16* __restrict__ qh, _Float16* __restrict__ kT,
              _Float16* __restrict__ vh,
              int M, int N, int K, int mode)
{
    const int lane  = threadIdx.x & 31;
    const int wave  = threadIdx.x >> 5;
    const int tile  = blockIdx.x * (blockDim.x >> 5) + wave;
    const int ntile = N >> 4;
    const int total = (M >> 4) * ntile;
    if (tile >= total) return;

    const int mt = tile / ntile, nt = tile % ntile;
    const int m0 = mt << 4, n0 = nt << 4;
    const int kb = (lane >> 4) << 3;
    // A frag: lane = M row, halves = K chunk (table order)
    const _Float16* ap = A + (size_t)(m0 + (lane & 15)) * K + kb;
    // B frag: lane%16 = N column, elements = K rows (table order)
    const _Float16* bp = Wt + n0 + (lane & 15);

    v8f acc = {};
    for (int k0 = 0; k0 < K; k0 += 32) {
        Frag a, b;
        a.h2[0] = *(const v8h*)(ap + k0);
        a.h2[1] = *(const v8h*)(ap + k0 + 16);
        const _Float16* bk = bp + (size_t)k0 * N;
        for (int s = 0; s < 16; ++s)
            b.e[s] = bk[(size_t)(kb + (s & 7) + ((s >> 3) << 4)) * N];
        __builtin_prefetch(ap + k0 + 64, 0, 1);   // global_prefetch_b8
        acc = __builtin_amdgcn_wmma_f32_16x16x32_f16(
            false, a.v, false, b.v, (short)0, acc, false, false);
    }

    // C/D layout: lane -> column n0 + lane%16; VGPR r -> row r + 8*(lane/16)
    const int col   = n0 + (lane & 15);
    const int rbase = (lane >> 4) << 3;
    const float bb  = bias[col];

    if (mode == 0) {
#pragma unroll
        for (int r = 0; r < 8; ++r)
            outF32[(size_t)(m0 + rbase + r) * N + col] = acc[r] + bb;
    } else {
        const int hh = col / 192;
        const int c  = col % 192;
#pragma unroll
        for (int r = 0; r < 8; ++r) {
            const int mm = m0 + rbase + r;
            const _Float16 val = (_Float16)(acc[r] + bb);
            if (c < KD) {
                qh[(size_t)mm * (NHEADS * KD) + hh * KD + c] = val;
            } else if (c < 2 * KD) {
                const int bt = mm / SEQ, tk = mm % SEQ;
                kT[((size_t)(bt * NHEADS + hh) * KD + (c - KD)) * KTP + tk] = val;
            } else {
                vh[(size_t)mm * DHID + hh * VD + (c - 2 * KD)] = val;
            }
        }
    }
}

// ---------------------------------------------------------------------------
// Fused attention: one 8-wave block per (batch, head).
// LDS: V (224x128 f16, padded keys zeroed) + K^T (32x208 f16) staged with
// async global->LDS loads; per-wave score slab (16x224 f32) with in-place
// f16 probs. Each wave owns q-tiles {wave, wave+8}.
// ---------------------------------------------------------------------------
#define SMEM_V_BYTES   (SEQP * VD * 2)            // 57344
#define SMEM_KT_BYTES  (KD * KTP * 2)             // 13312
#define SMEM_SC_OFF    (SMEM_V_BYTES + SMEM_KT_BYTES)   // 70656
#define SMEM_TOTAL     (SMEM_SC_OFF + 8 * 16 * SEQP * 4) // 185344

__global__ void __launch_bounds__(256)
attn_wmma(const _Float16* __restrict__ qh, const _Float16* __restrict__ kT,
          const _Float16* __restrict__ vh, const float* __restrict__ biases,
          const int* __restrict__ bidx, _Float16* __restrict__ ctx, int n_off)
{
    extern __shared__ char smem[];
    _Float16* sV  = (_Float16*)smem;                    // [224][128]
    _Float16* sKT = (_Float16*)(smem + SMEM_V_BYTES);   // [32][208]
    float*    sSc = (float*)(smem + SMEM_SC_OFF);       // [8][16][224]

    const int tid  = threadIdx.x;
    const int h    = blockIdx.x % NHEADS;
    const int b    = blockIdx.x / NHEADS;
    const int lane = tid & 31;
    const int wave = tid >> 5;
    const float scale = 0.17677669529663687f;   // 32^-0.5

    // ---- zero V padding rows (keys 196..223) ----
    for (int i = tid; i < (SEQP - SEQ) * VD; i += 256)
        sV[SEQ * VD + i] = (_Float16)0.f;

    // ---- async stage V: 196 rows x 128 halves (16B chunks) ----
    const _Float16* vsrc = vh + (size_t)b * SEQ * DHID + h * VD;
    for (int ci = tid; ci < SEQ * (VD / 8); ci += 256) {
        const int key = ci >> 4, c8 = (ci & 15) << 3;
        async_copy_b128(vsrc + (size_t)key * DHID + c8, sV + key * VD + c8);
    }
    // ---- async stage K^T strip: 32*208 halves contiguous ----
    const _Float16* ksrc = kT + (size_t)(b * NHEADS + h) * KD * KTP;
    for (int ci = tid; ci < KD * KTP / 8; ci += 256)
        async_copy_b128(ksrc + ci * 8, sKT + ci * 8);

    async_wait_all();
    __syncthreads();

    const int mloc  = lane & 15;
    const int kb    = (lane >> 4) << 3;
    const int rbase = (lane >> 4) << 3;
    float* mySc = sSc + wave * 16 * SEQP;

    for (int qt = wave; qt < QTILES; qt += 8) {
        // ---- Q fragment (A: lane = q row, halves = key-dim chunk) ----
        const int qg  = qt * 16 + mloc;
        const int qgc = (qg < SEQ) ? qg : (SEQ - 1);
        Frag aq;
        const _Float16* qp = qh + (size_t)(b * SEQ + qgc) * (NHEADS * KD) + h * KD;
        aq.h2[0] = *(const v8h*)(qp + kb);
        aq.h2[1] = *(const v8h*)(qp + kb + 16);

        // ---- scores: 13 key tiles, B frag = sKT column, k rows in table order ----
        for (int kt = 0; kt < QTILES; ++kt) {
            Frag bk;
            const _Float16* kp = sKT + kt * 16 + (lane & 15);
            for (int s = 0; s < 16; ++s)
                bk.e[s] = kp[(kb + (s & 7) + ((s >> 3) << 4)) * KTP];
            v8f c = {};
            c = __builtin_amdgcn_wmma_f32_16x16x32_f16(
                false, aq.v, false, bk.v, (short)0, c, false, false);

            const int kg = kt * 16 + (lane & 15);
#pragma unroll
            for (int r = 0; r < 8; ++r) {
                const int qr = qt * 16 + rbase + r;
                float s = c[r] * scale;
                if (qr < SEQ && kg < SEQ)
                    s += biases[h * n_off + bidx[qr * SEQ + kg]];
                mySc[(rbase + r) * SEQP + kt * 16 + (lane & 15)] = s;
            }
        }
        wave_lds_fence();

        // ---- softmax: one lane per row; in-place f32 -> f16 probs ----
        if (lane < 16) {
            float* rowp = mySc + lane * SEQP;
            float mx = -1e30f;
            for (int c = 0; c < SEQ; ++c) mx = fmaxf(mx, rowp[c]);
            float sum = 0.f;
            for (int c = 0; c < SEQ; ++c) {
                float e = __expf(rowp[c] - mx);
                rowp[c] = e;
                sum += e;
            }
            const float inv = 1.f / sum;
            _Float16* prow = (_Float16*)rowp;   // write idx 2c <= read idx 4c: safe
            for (int c = 0; c < SEQ; ++c) prow[c] = (_Float16)(rowp[c] * inv);
            for (int c = SEQ; c < SEQP; ++c) prow[c] = (_Float16)0.f;
        }
        wave_lds_fence();

        // ---- ctx = probs(16x224) @ V(224x128): 8 N-tiles x 7 K-steps ----
        const _Float16* prb = (const _Float16*)mySc;  // row stride 448 halves
        for (int vt = 0; vt < 8; ++vt) {
            v8f acc = {};
            for (int ks = 0; ks < 7; ++ks) {
                Frag ap2, bv;
                const _Float16* pp = prb + mloc * (SEQP * 2) + ks * 32 + kb;
                ap2.h2[0] = *(const v8h*)pp;
                ap2.h2[1] = *(const v8h*)(pp + 16);
                const _Float16* vp = sV + (ks * 32) * VD + vt * 16 + (lane & 15);
                for (int s = 0; s < 16; ++s)
                    bv.e[s] = vp[(kb + (s & 7) + ((s >> 3) << 4)) * VD];
                acc = __builtin_amdgcn_wmma_f32_16x16x32_f16(
                    false, ap2.v, false, bv.v, (short)0, acc, false, false);
            }
#pragma unroll
            for (int r = 0; r < 8; ++r) {
                const int qr = qt * 16 + rbase + r;
                if (qr < SEQ)
                    ctx[(size_t)(b * SEQ + qr) * DHID + h * VD + vt * 16 + (lane & 15)] =
                        (_Float16)acc[r];
            }
        }
    }
}

// ---------------------------------------------------------------------------
// Host launcher
// ---------------------------------------------------------------------------
extern "C" void kernel_launch(void* const* d_in, const int* in_sizes, int n_in,
                              void* d_out, int out_size, void* d_ws, size_t ws_size,
                              hipStream_t stream)
{
    const float* hidden  = (const float*)d_in[0];
    const float* norm_w  = (const float*)d_in[1];
    const float* norm_b  = (const float*)d_in[2];
    const float* qkv_w   = (const float*)d_in[3];
    const float* qkv_b   = (const float*)d_in[4];
    const float* proj_w  = (const float*)d_in[5];
    const float* proj_b  = (const float*)d_in[6];
    const float* attn_bi = (const float*)d_in[7];
    const int*   bidx    = (const int*)d_in[8];
    const int n_off = in_sizes[7] / NHEADS;

    char* ws = (char*)d_ws;
    size_t off = 0;
    auto carve = [&](size_t bytes) -> char* {
        char* p = ws + off;
        off = (off + bytes + 255) & ~(size_t)255;
        return p;
    };
    _Float16* xh   = (_Float16*)carve((size_t)TOKENS * DIMC * 2);        // normed x
    _Float16* qwt  = (_Float16*)carve((size_t)DIMC * HQKV * 2);          // qkv_w^T f16
    _Float16* pwt  = (_Float16*)carve((size_t)DHID * DIMC * 2);          // proj_w^T f16
    _Float16* qhB  = (_Float16*)carve((size_t)TOKENS * NHEADS * KD * 2); // q
    _Float16* kTB  = (_Float16*)carve((size_t)128 * NHEADS * KD * KTP * 2); // k^T
    _Float16* vhB  = (_Float16*)carve((size_t)TOKENS * DHID * 2);        // v
    _Float16* ctxh = (_Float16*)carve((size_t)TOKENS * DHID * 2);        // ctx
    (void)ws_size; (void)n_in; (void)out_size;

    // 1) weight convert + transpose to f16 [K][N]
    const int nqw = HQKV * DIMC;
    cvt_transpose_f16<<<(nqw + 255) / 256, 256, 0, stream>>>(qkv_w, qwt, HQKV, DIMC);
    const int npw = DIMC * DHID;
    cvt_transpose_f16<<<(npw + 255) / 256, 256, 0, stream>>>(proj_w, pwt, DIMC, DHID);

    // 2) layernorm -> f16
    layernorm_f16<<<TOKENS, 256, 0, stream>>>(hidden, norm_w, norm_b, xh);

    // 3) QKV GEMM with fused q/kT/v scatter
    const int tiles_qkv = (TOKENS / 16) * (HQKV / 16);   // 338688 -> 42336 blocks
    gemm_f16_wmma<<<tiles_qkv / 8, 256, 0, stream>>>(
        xh, qwt, qkv_b, nullptr, qhB, kTB, vhB, TOKENS, HQKV, DIMC, 1);

    // 4) fused attention: one block per (b, h), async-LDS staged K^T and V
    attn_wmma<<<128 * NHEADS, 256, SMEM_TOTAL, stream>>>(
        qhB, kTB, vhB, attn_bi, bidx, ctxh, n_off);

    // 5) projection GEMM -> f32 output
    const int tiles_p = (TOKENS / 16) * (DIMC / 16);     // 56448 -> 7056 blocks
    gemm_f16_wmma<<<tiles_p / 8, 256, 0, stream>>>(
        ctxh, pwt, proj_b, (float*)d_out, nullptr, nullptr, nullptr,
        TOKENS, DIMC, DHID, 0);
}
